// ChromaEncoder_53566832116024
// MI455X (gfx1250) — matrix operand order repaired and checked by reading, not compiled, not run on hardware
//
#include <hip/hip_runtime.h>
#include <hip/hip_bf16.h>
#include <math.h>

// ---------------- model constants (match reference setup) ----------------
#define BATCH   4
#define SEQ     1024
#define HID     1024
#define FFN     4096
#define LAT     512
#define NLAYERS 4
#define NCHROMA 12
#define NHEADS  16
#define HDIM    64
#define WIN     16
#define ROWS    (BATCH * SEQ)      // 4096 "M" rows for every GEMM
#define LN_EPS  1e-5f

typedef __attribute__((ext_vector_type(16))) _Float16 v16h;
typedef __attribute__((ext_vector_type(8)))  float    v8f;

union Frag { v16h v; _Float16 h[16]; unsigned u[8]; };
union Acc  { v8f  v; float f[8]; };

// epilogue flags
#define EPI_GELU  1
#define EPI_F32   2
#define EPI_F16   4

// ---- CDNA5 async global->LDS path (guarded; sync fallback keeps build safe) ----
#if defined(__has_builtin)
#  if __has_builtin(__builtin_amdgcn_global_load_async_to_lds_b128)
#    define GEMM_ASYNC 1
#  endif
#  if __has_builtin(__builtin_amdgcn_s_wait_asynccnt)
#    define HAVE_WAIT_ASYNC 1
#  endif
#endif

// builtin parameter types: pointer to 4 x i32 vector in addrspace(1)/(3)
typedef int v4i __attribute__((vector_size(16)));
typedef __attribute__((address_space(1))) v4i GInt4;   // global
typedef __attribute__((address_space(3))) v4i LInt4;   // LDS

template <int CNT>
__device__ __forceinline__ void wait_async_n() {
#if defined(GEMM_ASYNC)
#  if defined(HAVE_WAIT_ASYNC)
  __builtin_amdgcn_s_wait_asynccnt(CNT);
#  else
  asm volatile("s_wait_asynccnt %0" :: "i"(CNT) : "memory");
#  endif
#endif
}

// =====================================================================
// 1) Cantor routes: 16 nearest (stable ties) per row, block-per-row
// =====================================================================
__global__ __launch_bounds__(256) void routes_kernel(int* __restrict__ routes) {
  __shared__ double cs[SEQ];
  __shared__ double rd[256];
  __shared__ int    rj[256];
  __shared__ int    sel[WIN];
  const int i = blockIdx.x, tid = threadIdx.x;

  for (int j = tid; j < SEQ; j += 256) {
    double x = (double)j / (double)(SEQ - 1);
    x = fmin(fmax(x, 1e-6), 1.0 - 1e-6);
    double v = 0.0, f = 0.5;
    for (int d = 0; d < 8; ++d) {
      x *= 3.0; int dg = (int)x; x -= (double)dg;
      if (dg == 2) v += f;
      f *= 0.5;
    }
    cs[j] = v;
  }
  __syncthreads();
  const double ci = cs[i];

  for (int p = 0; p < WIN; ++p) {
    double best = 1e300; int bj = 1 << 30;
    for (int j = tid; j < SEQ; j += 256) {
      bool used = false;
      for (int q = 0; q < p; ++q) used |= (sel[q] == j);
      if (used) continue;
      double d = fabs(ci - cs[j]);
      if (d < best || (d == best && j < bj)) { best = d; bj = j; }
    }
    rd[tid] = best; rj[tid] = bj;
    __syncthreads();
    for (int s = 128; s > 0; s >>= 1) {
      if (tid < s) {
        if (rd[tid + s] < rd[tid] ||
            (rd[tid + s] == rd[tid] && rj[tid + s] < rj[tid])) {
          rd[tid] = rd[tid + s]; rj[tid] = rj[tid + s];
        }
      }
      __syncthreads();
    }
    if (tid == 0) { sel[p] = rj[0]; routes[i * WIN + p] = rj[0]; }
    __syncthreads();
  }
}

// =====================================================================
// 2) Embedding: x = chroma @ (W_emb + enc) + b_emb   (K=12, VALU)
// =====================================================================
__global__ __launch_bounds__(256) void embed_kernel(
    const float* __restrict__ chroma, const float* __restrict__ W_emb,
    const float* __restrict__ b_emb, float* __restrict__ x,
    _Float16* __restrict__ x16) {
  const int idx = blockIdx.x * 256 + threadIdx.x;   // < ROWS*HID
  const int row = idx >> 10, dd = idx & 1023;
  const float* ch = chroma + (size_t)row * NCHROMA;
  const int j = dd >> 1;
  const float base = 3.14159265358979323846f * ((float)(j + 1) / 512.0f) / 6.0f; // 2*pi*freq/12
  float acc = b_emb[dd];
  #pragma unroll
  for (int c = 0; c < NCHROMA; ++c) {
    float ang = base * (float)c;
    float e = (dd & 1) ? __sinf(ang) : __cosf(ang);
    acc += ch[c] * (W_emb[c * HID + dd] + e);
  }
  x[idx] = acc;
  x16[idx] = (_Float16)acc;
}

// =====================================================================
// 3) fused fp32->fp16 + transpose (weights K x N -> N x K), LDS-tiled
// =====================================================================
__global__ __launch_bounds__(256) void cvt_t_kernel(
    const float* __restrict__ in, _Float16* __restrict__ out, int K, int N) {
  __shared__ _Float16 t[32][33];
  const int kb = blockIdx.y * 32, nb = blockIdx.x * 32;
  const int tx = threadIdx.x & 31, ty = threadIdx.x >> 5;    // 32 x 8
  #pragma unroll
  for (int i = 0; i < 4; ++i) {
    const int k = kb + ty + i * 8, n = nb + tx;
    t[tx][ty + i * 8] = (_Float16)in[(size_t)k * N + n];
  }
  __syncthreads();
  #pragma unroll
  for (int i = 0; i < 4; ++i) {
    const int n = nb + ty + i * 8, k = kb + tx;
    out[(size_t)n * K + k] = t[ty + i * 8][tx];
  }
}

// =====================================================================
// 4) WMMA f16 GEMM: C[M,N] = A[M,K] * Bt[N,K]^T + bias, optional GELU
//    block tile 128x128, 8 waves, wave tile 32x64 (2x4 WMMA 16x16x32),
//    double-buffered LDS, async global->LDS prefetch when available
// =====================================================================
__device__ __forceinline__ float gelu_exact(float v) {
  return 0.5f * v * (1.0f + erff(v * 0.70710678118654752f));
}

// stage a 128x32 half tile (row-major, K-minor, ld = K) into LDS
__device__ __forceinline__ void stage_tile(const _Float16* __restrict__ g,
                                           int ld, int k0,
                                           _Float16 (&dst)[128][40], int tid) {
  #pragma unroll
  for (int i = 0; i < 2; ++i) {
    const int lin = tid + i * 256;                 // 0..511 chunks of 8 halves
    const int r = lin >> 2, c = (lin & 3) * 8;
    const _Float16* src = g + (size_t)r * ld + k0 + c;
#if defined(GEMM_ASYNC)
    __builtin_amdgcn_global_load_async_to_lds_b128(
        (GInt4*)src, (LInt4*)&dst[r][c], 0, 0);
#else
    *reinterpret_cast<uint4*>(&dst[r][c]) = *reinterpret_cast<const uint4*>(src);
#endif
  }
}

__global__ __launch_bounds__(256) void gemm_f16_kernel(
    const _Float16* __restrict__ A, const _Float16* __restrict__ Bt,
    const float* __restrict__ bias, float* __restrict__ Cf,
    _Float16* __restrict__ Ch, int M, int N, int K, int flags) {
  __shared__ __align__(16) _Float16 As[2][128][40];   // [buf][m][k]
  __shared__ __align__(16) _Float16 Bs[2][128][40];   // [buf][n][k]
  const int tid  = threadIdx.x;
  const int bm   = blockIdx.y * 128;
  const int bn   = blockIdx.x * 128;
  const int wave = tid >> 5, lane = tid & 31;
  const int wm   = (wave >> 1) * 32;                  // 0,32,64,96
  const int wn   = (wave & 1) * 64;                   // 0,64
  const int lmod = lane & 15, lhi = lane >> 4;

  Acc acc[2][4];
  #pragma unroll
  for (int i = 0; i < 2; ++i)
    #pragma unroll
    for (int j = 0; j < 4; ++j)
      #pragma unroll
      for (int r = 0; r < 8; ++r) acc[i][j].f[r] = 0.0f;

  const _Float16* Arow = A  + (size_t)bm * K;
  const _Float16* Brow = Bt + (size_t)bn * K;

  stage_tile(Arow, K, 0, As[0], tid);
  stage_tile(Brow, K, 0, Bs[0], tid);

  const int nk = K >> 5;
  for (int ks = 0; ks < nk; ++ks) {
    const int cur = ks & 1;
    if (ks + 1 < nk) {
      stage_tile(Arow, K, (ks + 1) << 5, As[cur ^ 1], tid);
      stage_tile(Brow, K, (ks + 1) << 5, Bs[cur ^ 1], tid);
      wait_async_n<4>();   // only the 4 just-issued (next-tile) ops may remain
    } else {
      wait_async_n<0>();
    }
    __syncthreads();

    // fragments per documented gfx1250 16-bit layouts
    Frag af[2], bf[4];
    #pragma unroll
    for (int s = 0; s < 2; ++s) {
      const int m = wm + s * 16 + lmod;
      #pragma unroll
      for (int r = 0; r < 8; ++r) {
        const int kf = ((r >> 2) << 4) + (lhi << 3) + ((r & 3) << 1);  // A layout
        af[s].u[r] = *reinterpret_cast<const unsigned*>(&As[cur][m][kf]);
      }
    }
    #pragma unroll
    for (int s = 0; s < 4; ++s) {
      const int n = wn + s * 16 + lmod;
      #pragma unroll
      for (int r = 0; r < 8; ++r) {
        const int kf = ((r >> 2) << 4) + (lhi << 3) + ((r & 3) << 1);  // B layout
        bf[s].u[r] = *reinterpret_cast<const unsigned*>(&Bs[cur][n][kf]);
      }
    }
    #pragma unroll
    for (int i = 0; i < 2; ++i)
      #pragma unroll
      for (int j = 0; j < 4; ++j)
        acc[i][j].v = __builtin_amdgcn_wmma_f32_16x16x32_f16(
            false, af[i].v, false, bf[j].v, (short)0, acc[i][j].v, false, false);
    __syncthreads();
  }

  // epilogue: C/D layout -> row = r + (lane>>4)*8, col = lane&15
  #pragma unroll
  for (int i = 0; i < 2; ++i)
    #pragma unroll
    for (int j = 0; j < 4; ++j) {
      const int col = bn + wn + j * 16 + lmod;
      const float bv = bias ? bias[col] : 0.0f;
      #pragma unroll
      for (int r = 0; r < 8; ++r) {
        const int row = bm + wm + i * 16 + (lhi << 3) + r;
        float v = acc[i][j].f[r] + bv;
        if (flags & EPI_GELU) v = gelu_exact(v);
        const size_t o = (size_t)row * N + col;
        if (flags & EPI_F32) Cf[o] = v;
        if (flags & EPI_F16) Ch[o] = (_Float16)v;
      }
    }
}

// =====================================================================
// 5) Windowed Cantor attention (fp32 VALU; ~0.1% of total FLOPs)
// =====================================================================
__global__ __launch_bounds__(256) void attn_kernel(
    const float* __restrict__ q, const float* __restrict__ k,
    const float* __restrict__ v, const int* __restrict__ routes,
    const float* __restrict__ temp, int layer, _Float16* __restrict__ out) {
  __shared__ int   rts[WIN];
  __shared__ float sc[NHEADS][WIN];
  __shared__ float at[NHEADS][WIN + 1];
  const int row = blockIdx.x;                 // 0..ROWS-1
  const int t = row & (SEQ - 1), b = row >> 10;
  const int tid = threadIdx.x;
  if (tid < WIN) rts[tid] = routes[t * WIN + tid];
  __syncthreads();

  const int h = tid >> 4, w = tid & 15;
  {
    const float* qr = q + (size_t)row * HID + h * HDIM;
    const float* kr = k + ((size_t)(b * SEQ + rts[w])) * HID + h * HDIM;
    float s = 0.0f;
    #pragma unroll 8
    for (int j = 0; j < HDIM; ++j) s += qr[j] * kr[j];
    sc[h][w] = s * (1.0f / (8.0f * fabsf(temp[layer])));
  }
  __syncthreads();
  {
    float mx = -1e30f;
    #pragma unroll
    for (int i = 0; i < WIN; ++i) mx = fmaxf(mx, sc[h][i]);
    float sum = 0.0f;
    #pragma unroll
    for (int i = 0; i < WIN; ++i) sum += expf(sc[h][i] - mx);
    at[h][w] = expf(sc[h][w] - mx) / sum;
  }
  __syncthreads();
  #pragma unroll
  for (int i = 0; i < 4; ++i) {
    const int dim = tid + i * 256;
    const int hh = dim >> 6;
    float a = 0.0f;
    #pragma unroll
    for (int ww = 0; ww < WIN; ++ww)
      a += at[hh][ww] * v[((size_t)(b * SEQ + rts[ww])) * HID + dim];
    out[(size_t)row * HID + dim] = (_Float16)a;
  }
}

// =====================================================================
// 6) residual add + layernorm (in-place on x_f32, also emits f16)
// =====================================================================
__global__ __launch_bounds__(256) void add_ln_kernel(
    float* __restrict__ x, const float* __restrict__ delta,
    const float* __restrict__ g, const float* __restrict__ bta,
    _Float16* __restrict__ x16) {
  __shared__ float r1[256], r2[256];
  const int row = blockIdx.x, tid = threadIdx.x;
  const size_t base = (size_t)row * HID;
  float s = 0.0f, s2 = 0.0f;
  for (int i = tid; i < HID; i += 256) {
    float y = x[base + i] + delta[base + i];
    s += y; s2 += y * y;
  }
  r1[tid] = s; r2[tid] = s2;
  __syncthreads();
  for (int st = 128; st > 0; st >>= 1) {
    if (tid < st) { r1[tid] += r1[tid + st]; r2[tid] += r2[tid + st]; }
    __syncthreads();
  }
  const float m = r1[0] * (1.0f / HID);
  const float var = r2[0] * (1.0f / HID) - m * m;
  const float inv = 1.0f / sqrtf(var + LN_EPS);
  for (int i = tid; i < HID; i += 256) {
    float y = x[base + i] + delta[base + i];
    float o = (y - m) * inv * g[i] + bta[i];
    x[base + i] = o;
    x16[base + i] = (_Float16)o;
  }
}

// =====================================================================
// driver
// =====================================================================
static inline void run_gemm(const _Float16* A, const _Float16* Bt, const float* bias,
                            float* Cf, _Float16* Ch, int M, int N, int K, int flags,
                            hipStream_t stream) {
  dim3 grid(N / 128, M / 128, 1);
  gemm_f16_kernel<<<grid, 256, 0, stream>>>(A, Bt, bias, Cf, Ch, M, N, K, flags);
}

extern "C" void kernel_launch(void* const* d_in, const int* in_sizes, int n_in,
                              void* d_out, int out_size, void* d_ws, size_t ws_size,
                              hipStream_t stream) {
  // ---- inputs (all fp32, setup_inputs order) ----
  const float* chroma = (const float*)d_in[0];
  const float* W_emb  = (const float*)d_in[1];
  const float* b_emb  = (const float*)d_in[2];
  const float* Wq     = (const float*)d_in[3];
  const float* bq     = (const float*)d_in[4];
  const float* Wk     = (const float*)d_in[5];
  const float* bk     = (const float*)d_in[6];
  const float* Wv     = (const float*)d_in[7];
  const float* bv     = (const float*)d_in[8];
  const float* Wo     = (const float*)d_in[9];
  const float* bo     = (const float*)d_in[10];
  const float* temp   = (const float*)d_in[11];
  const float* ln_g   = (const float*)d_in[12];
  const float* ln_b   = (const float*)d_in[13];
  const float* W1     = (const float*)d_in[14];
  const float* b1     = (const float*)d_in[15];
  const float* W2     = (const float*)d_in[16];
  const float* b2     = (const float*)d_in[17];
  const float* Wmu    = (const float*)d_in[18];
  const float* bmu    = (const float*)d_in[19];
  const float* Wlv    = (const float*)d_in[20];
  const float* blv    = (const float*)d_in[21];

  // ---- workspace carve-out (~226 MB) ----
  char* ws = (char*)d_ws;
  size_t off = 0;
  auto take = [&](size_t bytes) -> void* {
    off = (off + 255) & ~(size_t)255;
    void* p = ws + off;
    off += bytes;
    return p;
  };
  int*      routes = (int*)      take((size_t)SEQ * WIN * 4);
  float*    x_f32  = (float*)    take((size_t)ROWS * HID * 4);
  _Float16* x_f16  = (_Float16*) take((size_t)ROWS * HID * 2);
  float*    q_f32  = (float*)    take((size_t)ROWS * HID * 4);
  float*    k_f32  = (float*)    take((size_t)ROWS * HID * 4);
  float*    v_f32  = (float*)    take((size_t)ROWS * HID * 4);
  _Float16* a_f16  = (_Float16*) take((size_t)ROWS * HID * 2);
  float*    t_f32  = (float*)    take((size_t)ROWS * HID * 4);
  _Float16* h_f16  = (_Float16*) take((size_t)ROWS * FFN * 2);
  _Float16* wq16   = (_Float16*) take((size_t)NLAYERS * HID * HID * 2);  // transposed N x K
  _Float16* wk16   = (_Float16*) take((size_t)NLAYERS * HID * HID * 2);
  _Float16* wv16   = (_Float16*) take((size_t)NLAYERS * HID * HID * 2);
  _Float16* wo16   = (_Float16*) take((size_t)NLAYERS * HID * HID * 2);
  _Float16* w116   = (_Float16*) take((size_t)NLAYERS * HID * FFN * 2);  // FFN x HID
  _Float16* w216   = (_Float16*) take((size_t)NLAYERS * FFN * HID * 2);  // HID x FFN
  _Float16* wmu16  = (_Float16*) take((size_t)HID * LAT * 2);            // LAT x HID
  _Float16* wlv16  = (_Float16*) take((size_t)HID * LAT * 2);
  (void)ws_size; (void)n_in; (void)in_sizes; (void)out_size;

  // ---- precompute routes + embedding ----
  routes_kernel<<<SEQ, 256, 0, stream>>>(routes);
  embed_kernel<<<(ROWS * HID) / 256, 256, 0, stream>>>(chroma, W_emb, b_emb, x_f32, x_f16);

  // ---- convert + transpose weights to f16 N-major once per call ----
  auto cvt_t = [&](const float* in, _Float16* out, int K, int N) {
    dim3 grid(N / 32, K / 32, 1);
    cvt_t_kernel<<<grid, 256, 0, stream>>>(in, out, K, N);
  };
  for (int l = 0; l < NLAYERS; ++l) {
    const size_t o1 = (size_t)l * HID * HID;
    cvt_t(Wq + o1, wq16 + o1, HID, HID);
    cvt_t(Wk + o1, wk16 + o1, HID, HID);
    cvt_t(Wv + o1, wv16 + o1, HID, HID);
    cvt_t(Wo + o1, wo16 + o1, HID, HID);
    cvt_t(W1 + (size_t)l * HID * FFN, w116 + (size_t)l * HID * FFN, HID, FFN);
    cvt_t(W2 + (size_t)l * FFN * HID, w216 + (size_t)l * FFN * HID, FFN, HID);
  }
  cvt_t(Wmu, wmu16, HID, LAT);
  cvt_t(Wlv, wlv16, HID, LAT);

  // ---- transformer layers ----
  for (int l = 0; l < NLAYERS; ++l) {
    const size_t wOff = (size_t)l * HID * HID;
    run_gemm(x_f16, wq16 + wOff, bq + l * HID, q_f32, nullptr, ROWS, HID, HID, EPI_F32, stream);
    run_gemm(x_f16, wk16 + wOff, bk + l * HID, k_f32, nullptr, ROWS, HID, HID, EPI_F32, stream);
    run_gemm(x_f16, wv16 + wOff, bv + l * HID, v_f32, nullptr, ROWS, HID, HID, EPI_F32, stream);
    attn_kernel<<<ROWS, 256, 0, stream>>>(q_f32, k_f32, v_f32, routes, temp, l, a_f16);
    run_gemm(a_f16, wo16 + wOff, bo + l * HID, t_f32, nullptr, ROWS, HID, HID, EPI_F32, stream);
    add_ln_kernel<<<ROWS, 256, 0, stream>>>(x_f32, t_f32, ln_g + l * HID, ln_b + l * HID, x_f16);
    run_gemm(x_f16, w116 + (size_t)l * HID * FFN, b1 + l * FFN, nullptr, h_f16,
             ROWS, FFN, HID, EPI_F16 | EPI_GELU, stream);
    run_gemm(h_f16, w216 + (size_t)l * FFN * HID, b2 + l * HID, t_f32, nullptr,
             ROWS, HID, FFN, EPI_F32, stream);
    add_ln_kernel<<<ROWS, 256, 0, stream>>>(x_f32, t_f32, ln_g + l * HID, ln_b + l * HID, x_f16);
  }

  // ---- heads: mu then lv, fp32 straight into d_out ----
  float* outp = (float*)d_out;
  run_gemm(x_f16, wmu16, bmu, outp, nullptr, ROWS, LAT, HID, EPI_F32, stream);
  run_gemm(x_f16, wlv16, blv, outp + (size_t)ROWS * LAT, nullptr, ROWS, LAT, HID, EPI_F32, stream);
}
